// FactorizedSpatioTemporalGNS_heterogeneous_81810537054567
// MI455X (gfx1250) — matrix operand order repaired and checked by reading, not compiled, not run on hardware
//
#include <hip/hip_runtime.h>

// ---------------------------------------------------------------------------
// FactorizedSpatioTemporalGNS forward for MI455X (gfx1250, wave32, WMMA).
// All dense linears -> f16 WMMA GEMM (v_wmma_f32_16x16x32_f16, f32 accum),
// 128x128 block tile, 32x64 wave tile (8 WMMA / K-panel / wave), double-
// buffered LDS with GLOBAL_LOAD_ASYNC_TO_LDS for the f16 weight tiles and
// global_prefetch for lookahead. Graph attention -> 3-pass segment softmax.
// ---------------------------------------------------------------------------

#define NBUS 2000
#define NGEN 500
#define TT   24
#define EBB  8000
#define DH   512   // HID*HEADS
#define HIDC 128   // per-head channels

#if __has_builtin(__builtin_amdgcn_global_load_async_to_lds_b128) && \
    __has_builtin(__builtin_amdgcn_s_wait_asynccnt)
#define USE_ASYNC_LDS 1
#else
#define USE_ASYNC_LDS 0
#endif

typedef __attribute__((ext_vector_type(16))) _Float16 v16h;
typedef __attribute__((ext_vector_type(8)))  _Float16 v8h;
typedef __attribute__((ext_vector_type(8)))  float    v8f;
typedef __attribute__((ext_vector_type(4)))  float    v4f;
typedef __attribute__((ext_vector_type(4)))  int      v4i;
typedef __attribute__((address_space(1))) v4i* p_gv4i;   // global int4*
typedef __attribute__((address_space(3))) v4i* p_lv4i;   // LDS int4*

// ------------------------- small device helpers ----------------------------
__device__ __forceinline__ unsigned fkey(float f) {
  unsigned u = __float_as_uint(f);
  return (u & 0x80000000u) ? ~u : (u | 0x80000000u);
}
__device__ __forceinline__ float fkeyinv(unsigned k) {
  unsigned u = (k & 0x80000000u) ? (k ^ 0x80000000u) : ~k;
  return __uint_as_float(u);
}

__global__ __launch_bounds__(256) void k_fill_f32(float* p, float v, size_t n) {
  size_t g = (size_t)blockIdx.x * 256 + threadIdx.x;
  if (g < n) p[g] = v;
}
__global__ __launch_bounds__(256) void k_fill_u32(unsigned* p, unsigned v, size_t n) {
  size_t g = (size_t)blockIdx.x * 256 + threadIdx.x;
  if (g < n) p[g] = v;
}

// W[K,N] f32 -> Wt[N,K] f16 (so GEMM B-tiles are K-contiguous in LDS)
__global__ __launch_bounds__(256) void k_cvtw(const float* __restrict__ w,
                                              _Float16* __restrict__ wt,
                                              int K, int N) {
  size_t gid = (size_t)blockIdx.x * 256 + threadIdx.x;
  if (gid >= (size_t)K * N) return;
  int kk = (int)(gid / N), n = (int)(gid % N);
  wt[(size_t)n * K + kk] = (_Float16)w[gid];
}

// ------------------------------ WMMA GEMM ----------------------------------
// C[M,N] = act(A[M,K](f32,lda) @ Bt[N,K](f16)^T + bias) + res
// Block 128x128, K-panel 32, 256 threads = 8 waves (4 M x 2 N), wave 32x64.
__global__ __launch_bounds__(256) void k_gemm(
    const float* __restrict__ A, int lda,
    const _Float16* __restrict__ Bt,
    const float* __restrict__ bias, const float* __restrict__ res,
    float* __restrict__ C, int M, int N, int K, int act) {
  __shared__ _Float16 As[2][128][40];   // row stride 80B -> 16B aligned chunks
  __shared__ _Float16 Bs[2][128][40];

  const int tid  = threadIdx.x;
  const int lane = tid & 31;
  const int wave = tid >> 5;
  const int wm = wave & 3;              // M quarter (0..3) -> 32 rows
  const int wn = wave >> 2;             // N half    (0..1) -> 64 cols
  const int m0 = blockIdx.y * 128;
  const int n0 = blockIdx.x * 128;

  // staging coordinates: each thread owns 16 contiguous K-elems of one row
  const int sr = tid >> 1;              // 0..127
  const int sc = (tid & 1) * 16;        // 0,16

  v8f acc[2][4];
#pragma unroll
  for (int mi = 0; mi < 2; ++mi)
#pragma unroll
    for (int ni = 0; ni < 4; ++ni)
#pragma unroll
      for (int i = 0; i < 8; ++i) acc[mi][ni][i] = 0.f;

  float areg[16];

  // ---- stage helpers -------------------------------------------------------
  auto stageA_load = [&](int kn) {
    int gr = m0 + sr;
    if (gr < M && kn + sc + 16 <= K) {
      const float* ap = A + (size_t)gr * lda + kn + sc;
#pragma unroll
      for (int q = 0; q < 4; ++q) {
        v4f x = *(const v4f*)(ap + q * 4);
#pragma unroll
        for (int i = 0; i < 4; ++i) areg[q * 4 + i] = x[i];
      }
      if (kn + sc + 48 <= K) __builtin_prefetch(ap + 32, 0, 0);  // 2 panels ahead
    } else {
#pragma unroll
      for (int i = 0; i < 16; ++i) {
        int kk = kn + sc + i;
        areg[i] = (gr < M && kk < K) ? A[(size_t)gr * lda + kk] : 0.f;
      }
    }
  };
  auto stageA_store = [&](int buf) {
    _Float16 h[16];
#pragma unroll
    for (int i = 0; i < 16; ++i) h[i] = (_Float16)areg[i];
    _Float16* dp = &As[buf][sr][sc];
    *(v8h*)dp = *(const v8h*)&h[0];
    *(v8h*)(dp + 8) = *(const v8h*)&h[8];
  };
  auto stageB = [&](int kn, int buf) {
    int gn = n0 + sr;
    _Float16* dp = &Bs[buf][sr][sc];
    if (gn < N && kn + sc + 16 <= K) {
      const _Float16* bp = Bt + (size_t)gn * K + kn + sc;
#if USE_ASYNC_LDS
      __builtin_amdgcn_global_load_async_to_lds_b128(
          (p_gv4i)(void*)bp, (p_lv4i)(void*)dp, 0, 0);
      __builtin_amdgcn_global_load_async_to_lds_b128(
          (p_gv4i)(void*)(bp + 8), (p_lv4i)(void*)(dp + 8), 0, 0);
#else
      *(v8h*)dp       = *(const v8h*)bp;
      *(v8h*)(dp + 8) = *(const v8h*)(bp + 8);
#endif
      if (kn + sc + 48 <= K) __builtin_prefetch(bp + 32, 0, 0);
    } else {
#pragma unroll
      for (int i = 0; i < 16; ++i) {
        int kk = kn + sc + i;
        dp[i] = (gn < N && kk < K) ? Bt[(size_t)gn * K + kk] : (_Float16)0.f;
      }
    }
  };

  // ---- prologue: panel 0 into buffer 0 ----
  stageA_load(0);
  stageB(0, 0);
  stageA_store(0);
#if USE_ASYNC_LDS
  __builtin_amdgcn_s_wait_asynccnt(0);
#endif
  __syncthreads();

  int buf = 0;
  for (int k0 = 0; k0 < K; k0 += 32) {
    const int kn = k0 + 32;
    const bool more = kn < K;
    if (more) { stageA_load(kn); stageB(kn, buf ^ 1); }

    // ---- fragments per ISA 7.12.2 layouts ----
    const int am  = lane & 15;
    const int akb = (lane >> 4) * 8;        // A: K base {0,8}; +16 upper half
    v16h af[2];
#pragma unroll
    for (int mi = 0; mi < 2; ++mi) {
      const _Float16* p = &As[buf][wm * 32 + mi * 16 + am][0];
      v8h lo = *(const v8h*)(p + akb);
      v8h hi = *(const v8h*)(p + akb + 16);
      v16h t;
#pragma unroll
      for (int i = 0; i < 8; ++i) { t[i] = lo[i]; t[i + 8] = hi[i]; }
      af[mi] = t;
    }
    const int bn  = lane & 15;
    const int bkb = (lane >> 4) * 8;        // B: K base {0,8}; +16 upper half
    v16h bf[4];
#pragma unroll
    for (int ni = 0; ni < 4; ++ni) {
      const _Float16* p = &Bs[buf][wn * 64 + ni * 16 + bn][0];
      v8h lo = *(const v8h*)(p + bkb);
      v8h hi = *(const v8h*)(p + bkb + 16);
      v16h t;
#pragma unroll
      for (int i = 0; i < 8; ++i) { t[i] = lo[i]; t[i + 8] = hi[i]; }
      bf[ni] = t;
    }
#pragma unroll
    for (int mi = 0; mi < 2; ++mi)
#pragma unroll
      for (int ni = 0; ni < 4; ++ni)
        acc[mi][ni] = __builtin_amdgcn_wmma_f32_16x16x32_f16(
            false, af[mi], false, bf[ni], (short)0, acc[mi][ni], false, false);

    if (more) stageA_store(buf ^ 1);
#if USE_ASYNC_LDS
    __builtin_amdgcn_s_wait_asynccnt(0);
#endif
    __syncthreads();
    buf ^= 1;
  }

  // ---- epilogue: C/D layout lane l -> n=l&15, rows r + 8*(l>>4) ----
  const int cn  = lane & 15;
  const int cmh = (lane >> 4) * 8;
#pragma unroll
  for (int mi = 0; mi < 2; ++mi)
#pragma unroll
    for (int ni = 0; ni < 4; ++ni)
#pragma unroll
      for (int r = 0; r < 8; ++r) {
        int gm = m0 + wm * 32 + mi * 16 + cmh + r;
        int gn = n0 + wn * 64 + ni * 16 + cn;
        if (gm < M && gn < N) {
          float x = acc[mi][ni][r];
          if (bias) x += bias[gn];
          if (act == 1) x = x > 0.f ? x : 0.01f * x;      // leaky_relu
          else if (act == 2) x = fmaxf(x, 0.f);           // relu
          if (res) x += res[(size_t)gm * N + gn];
          C[(size_t)gm * N + gn] = x;
        }
      }
}

// --------------------------- layernorm (wave/row) --------------------------
__global__ __launch_bounds__(256) void k_ln(const float* __restrict__ x,
                                            const float* __restrict__ g,
                                            const float* __restrict__ b,
                                            const float* __restrict__ res,
                                            float* __restrict__ out,
                                            int rows, int dim, int act) {
  int row = blockIdx.x * 8 + (threadIdx.x >> 5);
  if (row >= rows) return;
  int lane = threadIdx.x & 31;
  const float* xp = x + (size_t)row * dim;
  float s = 0.f, s2 = 0.f;
  for (int j = lane; j < dim; j += 32) { float v = xp[j]; s += v; s2 += v * v; }
  for (int m = 16; m; m >>= 1) { s += __shfl_xor(s, m); s2 += __shfl_xor(s2, m); }
  float inv = 1.f / (float)dim;
  float mu = s * inv;
  float var = s2 * inv - mu * mu;
  float r = rsqrtf(var + 1e-5f);
  float* op = out + (size_t)row * dim;
  const float* rp = res ? res + (size_t)row * dim : nullptr;
  for (int j = lane; j < dim; j += 32) {
    float y = (xp[j] - mu) * r * g[j] + b[j];
    if (act == 1) y = y > 0.f ? y : 0.01f * y;
    if (rp) y += rp[j];
    op[j] = y;
  }
}

// ------------------------ positional encoding add --------------------------
__global__ __launch_bounds__(256) void k_pe(float* __restrict__ h, size_t rows) {
  size_t gid = (size_t)blockIdx.x * 256 + threadIdx.x;
  if (gid >= rows * 128) return;
  int c = (int)(gid & 127);
  int t = (int)((gid >> 7) % TT);
  int i2 = c & ~1;
  float div = __expf(-(float)i2 * (9.2103403719761836f / 128.f)); // ln(1e4)/128
  float ang = (float)t * div;
  h[gid] += (c & 1) ? __cosf(ang) : __sinf(ang);
}

// -------------------- graph attention (segment softmax) --------------------
__global__ __launch_bounds__(256) void k_edge_logits(
    const float* __restrict__ q, const float* __restrict__ k,
    const float* __restrict__ el,
    const int* __restrict__ src, const int* __restrict__ dst,
    int E, float* __restrict__ lg, unsigned* __restrict__ mkey) {
  int idx = blockIdx.x * 8 + (threadIdx.x >> 5);
  if (idx >= E * TT * 4) return;
  int lane = threadIdx.x & 31;
  int h = idx & 3, t = (idx >> 2) % TT, e = idx / (TT * 4);
  int s = src[e], d = dst[e];
  const float* qp = q + ((size_t)d * TT + t) * DH + h * HIDC;
  const float* kp = k + ((size_t)s * TT + t) * DH + h * HIDC;
  const float* ep = el ? el + ((size_t)e * TT + t) * DH + h * HIDC : nullptr;
  float sum = 0.f;
  for (int c = lane; c < HIDC; c += 32) {
    float kv = kp[c] + (ep ? ep[c] : 0.f);
    sum += qp[c] * kv;
  }
  for (int m = 16; m; m >>= 1) sum += __shfl_xor(sum, m);
  float a = sum * 0.0883883476483184f; // 1/sqrt(128)
  if (lane == 0) {
    lg[idx] = a;
    atomicMax(&mkey[((size_t)d * TT + t) * 4 + h], fkey(a));
  }
}

__global__ __launch_bounds__(256) void k_edge_den(
    const float* __restrict__ lg, const unsigned* __restrict__ mkey,
    const int* __restrict__ dst, int E, float* __restrict__ den) {
  int idx = blockIdx.x * 256 + threadIdx.x;
  if (idx >= E * TT * 4) return;
  int h = idx & 3, t = (idx >> 2) % TT, e = idx / (TT * 4);
  size_t di = ((size_t)dst[e] * TT + t) * 4 + h;
  atomicAdd(&den[di], __expf(lg[idx] - fkeyinv(mkey[di])));
}

__global__ __launch_bounds__(256) void k_edge_scatter(
    const float* __restrict__ lg, const unsigned* __restrict__ mkey,
    const float* __restrict__ den, const float* __restrict__ v,
    const float* __restrict__ el, const int* __restrict__ src,
    const int* __restrict__ dst, int E, float* __restrict__ out) {
  size_t gid = (size_t)blockIdx.x * 256 + threadIdx.x;
  size_t tot = (size_t)E * TT * 4 * HIDC;
  if (gid >= tot) return;
  int c = (int)(gid & 127);
  int idx = (int)(gid >> 7);
  int h = idx & 3, t = (idx >> 2) % TT, e = idx / (TT * 4);
  int s = src[e], d = dst[e];
  size_t di = ((size_t)d * TT + t) * 4 + h;
  float w = __expf(lg[idx] - fkeyinv(mkey[di])) / den[di];
  float val = v[((size_t)s * TT + t) * DH + h * HIDC + c];
  if (el) val += el[((size_t)e * TT + t) * DH + h * HIDC + c];
  atomicAdd(&out[((size_t)d * TT + t) * DH + h * HIDC + c], w * val);
}

// beta-gated combine: out = beta*xr + (1-beta)*aout  (optionally accumulate)
__global__ __launch_bounds__(256) void k_beta(
    const float* __restrict__ aout, const float* __restrict__ xr,
    const float* __restrict__ wb, const float* __restrict__ bb,
    float* __restrict__ out, int rows, int accum) {
  int row = blockIdx.x * 8 + (threadIdx.x >> 5);
  if (row >= rows) return;
  int lane = threadIdx.x & 31;
  const float* o = aout + (size_t)row * DH;
  const float* x = xr + (size_t)row * DH;
  float s = 0.f;
  for (int j = lane; j < DH; j += 32) {
    float ov = o[j], xv = x[j];
    s += wb[j] * ov + wb[DH + j] * xv + wb[2 * DH + j] * (ov - xv);
  }
  for (int m = 16; m; m >>= 1) s += __shfl_xor(s, m);
  float beta = 1.f / (1.f + __expf(-(s + bb[0])));
  float* op = out + (size_t)row * DH;
  for (int j = lane; j < DH; j += 32) {
    float r = beta * x[j] + (1.f - beta) * o[j];
    op[j] = accum ? op[j] + r : r;
  }
}

// ----------------------- temporal attention (T=24) -------------------------
__global__ __launch_bounds__(256) void k_tattn(const float* __restrict__ q,
                                               const float* __restrict__ k,
                                               const float* __restrict__ v,
                                               float* __restrict__ o) {
  __shared__ float qs[TT][HIDC], ks[TT][HIDC], vs[TT][HIDC];
  __shared__ float sc[TT][TT + 1];
  int node = blockIdx.x >> 2, h = blockIdx.x & 3;
  int tid = threadIdx.x;
  for (int i = tid; i < TT * HIDC; i += 256) {
    int t = i >> 7, c = i & 127;
    size_t r = ((size_t)node * TT + t) * DH + h * HIDC + c;
    qs[t][c] = q[r]; ks[t][c] = k[r]; vs[t][c] = v[r];
  }
  __syncthreads();
  for (int p = tid; p < TT * TT; p += 256) {
    int t = p / TT, s = p % TT;
    float a = 0.f;
    for (int c = 0; c < HIDC; ++c) a += qs[t][c] * ks[s][c];
    sc[t][s] = a * 0.0883883476483184f;
  }
  __syncthreads();
  if (tid < TT) {
    float m = -1e30f;
    for (int s = 0; s < TT; ++s) m = fmaxf(m, sc[tid][s]);
    float den = 0.f;
    for (int s = 0; s < TT; ++s) { float e = __expf(sc[tid][s] - m); sc[tid][s] = e; den += e; }
    float r = 1.f / den;
    for (int s = 0; s < TT; ++s) sc[tid][s] *= r;
  }
  __syncthreads();
  for (int i = tid; i < TT * HIDC; i += 256) {
    int t = i >> 7, c = i & 127;
    float a = 0.f;
    for (int s = 0; s < TT; ++s) a += sc[t][s] * vs[s][c];
    o[((size_t)node * TT + t) * DH + h * HIDC + c] = a;
  }
}

// ------------------------------- host side ---------------------------------
struct LinP { const float* b; const float* w; };
struct LNPp { const float* b; const float* g; };
struct ProjP { LinP l1, l2; LNPp ln; };
struct ConvP { LinP beta, e, k, q, skip, v; };
struct TempP { LinP f1, f2, k, o, q, v; LNPp ln1, ln2; };
struct LayerP { ConvP bb, bg, gb; LNPp ln_bus, ln_gen; TempP tb, tg; };

static inline float* carve(char*& p, size_t nfloats) {
  float* r = (float*)p;
  p += ((nfloats * 4 + 255) / 256) * 256;
  return r;
}

static void gemm(hipStream_t st, const float* A, int lda, const float* W,
                 const float* bias, const float* res, float* C,
                 int M, int N, int K, int act, _Float16* wt) {
  size_t wn = (size_t)K * N;
  k_cvtw<<<(unsigned)((wn + 255) / 256), 256, 0, st>>>(W, wt, K, N);
  dim3 grid((N + 127) / 128, (M + 127) / 128);
  k_gemm<<<grid, 256, 0, st>>>(A, lda, wt, bias, res, C, M, N, K, act);
}

static void run_proj(hipStream_t st, const ProjP& p, const float* x, int M,
                     int fin, float* t1, float* t2, float* out, _Float16* wt) {
  gemm(st, x, fin, p.l1.w, p.l1.b, nullptr, t1, M, HIDC, fin, 1, wt);
  gemm(st, t1, HIDC, p.l2.w, p.l2.b, nullptr, t2, M, HIDC, HIDC, 0, wt);
  k_ln<<<(M + 7) / 8, 256, 0, st>>>(t2, p.ln.g, p.ln.b, nullptr, out, M, HIDC, 0);
}

static void run_conv(hipStream_t st, const ConvP& cv,
                     const float* xsrc, int nsrc, const float* xdst, int ndst,
                     int fin, const int* sidx, const int* didx, int E,
                     const float* eproj, float* out, int accum,
                     float* Q, float* K, float* V, float* EL, float* LG,
                     unsigned* MK, float* DEN, float* AO, _Float16* WT) {
  int Md = ndst * TT, Ms = nsrc * TT;
  gemm(st, xdst, fin, cv.q.w, cv.q.b, nullptr, Q, Md, DH, fin, 0, WT);
  gemm(st, xsrc, fin, cv.k.w, cv.k.b, nullptr, K, Ms, DH, fin, 0, WT);
  const float* el = nullptr;
  if (eproj) {
    gemm(st, eproj, HIDC, cv.e.w, cv.e.b, nullptr, EL, E * TT, DH, HIDC, 0, WT);
    el = EL;
  }
  size_t nd4 = (size_t)ndst * TT * 4;
  k_fill_u32<<<(unsigned)((nd4 + 255) / 256), 256, 0, st>>>(MK, 0x007FFFFFu, nd4); // key(-inf)
  k_fill_f32<<<(unsigned)((nd4 + 255) / 256), 256, 0, st>>>(DEN, 0.f, nd4);
  size_t ao = (size_t)Md * DH;
  k_fill_f32<<<(unsigned)((ao + 255) / 256), 256, 0, st>>>(AO, 0.f, ao);
  int nl = E * TT * 4;
  k_edge_logits<<<(nl + 7) / 8, 256, 0, st>>>(Q, K, el, sidx, didx, E, LG, MK);
  k_edge_den<<<(nl + 255) / 256, 256, 0, st>>>(LG, MK, didx, E, DEN);
  gemm(st, xsrc, fin, cv.v.w, cv.v.b, nullptr, V, Ms, DH, fin, 0, WT);
  size_t ns = (size_t)nl * HIDC;
  k_edge_scatter<<<(unsigned)((ns + 255) / 256), 256, 0, st>>>(LG, MK, DEN, V, el, sidx, didx, E, AO);
  gemm(st, xdst, fin, cv.skip.w, cv.skip.b, nullptr, Q, Md, DH, fin, 0, WT); // skip reuses Q
  k_beta<<<(Md + 7) / 8, 256, 0, st>>>(AO, Q, cv.beta.w, cv.beta.b, out, Md, accum);
}

static void run_temporal(hipStream_t st, const TempP& tp, float* X, int n,
                         float* Q, float* K, float* V, float* AO, float* OB,
                         float* BIG, _Float16* WT) {
  int M = n * TT;
  gemm(st, X, DH, tp.q.w, tp.q.b, nullptr, Q, M, DH, DH, 0, WT);
  gemm(st, X, DH, tp.k.w, tp.k.b, nullptr, K, M, DH, DH, 0, WT);
  gemm(st, X, DH, tp.v.w, tp.v.b, nullptr, V, M, DH, DH, 0, WT);
  k_tattn<<<n * 4, 256, 0, st>>>(Q, K, V, AO);
  gemm(st, AO, DH, tp.o.w, tp.o.b, X, OB, M, DH, DH, 0, WT);               // x + o-proj
  k_ln<<<(M + 7) / 8, 256, 0, st>>>(OB, tp.ln1.g, tp.ln1.b, nullptr, X, M, DH, 0);
  gemm(st, X, DH, tp.f1.w, tp.f1.b, nullptr, BIG, M, 4 * DH, DH, 2, WT);   // relu FFN
  gemm(st, BIG, 4 * DH, tp.f2.w, tp.f2.b, X, OB, M, DH, 4 * DH, 0, WT);    // x + ffn
  k_ln<<<(M + 7) / 8, 256, 0, st>>>(OB, tp.ln2.g, tp.ln2.b, nullptr, X, M, DH, 0);
}

extern "C" void kernel_launch(void* const* d_in, const int* in_sizes, int n_in,
                              void* d_out, int out_size, void* d_ws, size_t ws_size,
                              hipStream_t stream) {
  (void)in_sizes; (void)n_in; (void)out_size; (void)ws_size;
  const float* x_bus  = (const float*)d_in[0];
  const float* x_gen  = (const float*)d_in[1];
  const float* e_attr = (const float*)d_in[2];
  const int*   ei     = (const int*)d_in[3];
  const int*   gbs    = (const int*)d_in[4];
  const int*   gbd    = (const int*)d_in[5];

  // params flattened in JAX pytree order (sorted dict keys, depth-first)
  int cur = 6;
  auto nxt = [&]() { return (const float*)d_in[cur++]; };
  auto getLin = [&]() { LinP l; l.b = nxt(); l.w = nxt(); return l; };
  auto getLN  = [&]() { LNPp p; p.b = nxt(); p.g = nxt(); return p; };
  auto getProj = [&]() { ProjP p; p.l1 = getLin(); p.l2 = getLin(); p.ln = getLN(); return p; };
  auto getConv = [&](bool has_e) {
    ConvP c; c.beta = getLin();
    if (has_e) c.e = getLin(); else { c.e.b = nullptr; c.e.w = nullptr; }
    c.k = getLin(); c.q = getLin(); c.skip = getLin(); c.v = getLin(); return c;
  };
  auto getTemp = [&]() {
    TempP t; t.f1 = getLin(); t.f2 = getLin(); t.k = getLin();
    t.ln1 = getLN(); t.ln2 = getLN(); t.o = getLin(); t.q = getLin(); t.v = getLin();
    return t;
  };

  ProjP Pbus = getProj();   // 'in_bus'
  ProjP Pedge = getProj();  // 'in_edge'
  ProjP Pgen = getProj();   // 'in_gen'
  LayerP layers[3];
  for (int i = 0; i < 3; ++i) {     // 'layers': bb, bg, gb, ln_bus, ln_gen, tb, tg
    layers[i].bb = getConv(true);
    layers[i].bg = getConv(false);
    layers[i].gb = getConv(false);
    layers[i].ln_bus = getLN();
    layers[i].ln_gen = getLN();
    layers[i].tb = getTemp();
    layers[i].tg = getTemp();
  }
  ProjP MbusT = getProj();  // 'mlp_bus': l1, l2, ln
  ProjP MgenT = getProj();  // 'mlp_gen'

  // ---------------- workspace carve ----------------
  char* wp = (char*)d_ws;
  const size_t MB = (size_t)NBUS * TT;   // 48000
  const size_t MG = (size_t)NGEN * TT;   // 12000
  const size_t ME = (size_t)EBB * TT;    // 192000
  float* HB   = carve(wp, MB * DH);
  float* HG   = carve(wp, MG * DH);
  float* EP   = carve(wp, ME * HIDC);
  float* Qb   = carve(wp, MB * DH);
  float* Kb   = carve(wp, MB * DH);
  float* Vb   = carve(wp, MB * DH);
  float* AOUT = carve(wp, MB * DH);
  float* OB   = carve(wp, MB * DH);
  float* OGB  = carve(wp, MG * DH);
  float* BIG  = carve(wp, MB * 4 * DH);          // FFN hidden / edge-lin (same size)
  float* LG   = carve(wp, (size_t)EBB * TT * 4);
  float* DEN  = carve(wp, MB * 4);
  unsigned* MK = (unsigned*)carve(wp, MB * 4);
  _Float16* WT = (_Float16*)carve(wp, (size_t)(4 * DH) * DH / 2 + 64); // f16 [N,K] scratch

  // ---------------- input projections + posenc ----------------
  run_proj(stream, Pbus, x_bus, (int)MB, 16, Qb, Kb, HB, WT);
  k_pe<<<(unsigned)((MB * 128 + 255) / 256), 256, 0, stream>>>(HB, MB);
  run_proj(stream, Pgen, x_gen, (int)MG, 8, Qb, Kb, HG, WT);
  k_pe<<<(unsigned)((MG * 128 + 255) / 256), 256, 0, stream>>>(HG, MG);
  run_proj(stream, Pedge, e_attr, (int)ME, 4, Qb, Kb, EP, WT);

  // ---------------- layers ----------------
  const int* bbs = ei;
  const int* bbd = ei + EBB;
  int fin = HIDC;
  for (int L = 0; L < 3; ++L) {
    const LayerP& lp = layers[L];
    run_conv(stream, lp.bb, HB, NBUS, HB, NBUS, fin, bbs, bbd, EBB, EP, OB, 0,
             Qb, Kb, Vb, BIG, LG, MK, DEN, AOUT, WT);
    run_conv(stream, lp.gb, HG, NGEN, HB, NBUS, fin, gbs, gbd, NGEN, nullptr, OB, 1,
             Qb, Kb, Vb, BIG, LG, MK, DEN, AOUT, WT);
    run_conv(stream, lp.bg, HB, NBUS, HG, NGEN, fin, gbd, gbs, NGEN, nullptr, OGB, 0,
             Qb, Kb, Vb, BIG, LG, MK, DEN, AOUT, WT);
    k_ln<<<(unsigned)((MB + 7) / 8), 256, 0, stream>>>(
        OB, lp.ln_bus.g, lp.ln_bus.b, (fin == DH) ? HB : nullptr, HB, (int)MB, DH, 1);
    k_ln<<<(unsigned)((MG + 7) / 8), 256, 0, stream>>>(
        OGB, lp.ln_gen.g, lp.ln_gen.b, (fin == DH) ? HG : nullptr, HG, (int)MG, DH, 1);
    run_temporal(stream, lp.tb, HB, NBUS, Qb, Kb, Vb, AOUT, OB, BIG, WT);
    run_temporal(stream, lp.tg, HG, NGEN, Qb, Kb, Vb, AOUT, OB, BIG, WT);
    fin = DH;
  }

  // ---------------- output MLPs ----------------
  float* outb = (float*)d_out;
  gemm(stream, HB, DH, MbusT.l1.w, MbusT.l1.b, nullptr, Qb, (int)MB, HIDC, DH, 0, WT);
  k_ln<<<(unsigned)((MB + 7) / 8), 256, 0, stream>>>(Qb, MbusT.ln.g, MbusT.ln.b, nullptr, Kb, (int)MB, HIDC, 1);
  gemm(stream, Kb, HIDC, MbusT.l2.w, MbusT.l2.b, nullptr, outb, (int)MB, 4, HIDC, 0, WT);
  gemm(stream, HG, DH, MgenT.l1.w, MgenT.l1.b, nullptr, Qb, (int)MG, HIDC, DH, 0, WT);
  k_ln<<<(unsigned)((MG + 7) / 8), 256, 0, stream>>>(Qb, MgenT.ln.g, MgenT.ln.b, nullptr, Kb, (int)MG, HIDC, 1);
  gemm(stream, Kb, HIDC, MgenT.l2.w, MgenT.l2.b, nullptr, outb + MB * 4, (int)MG, 2, HIDC, 0, WT);
}
